// ConditionedBlockInvariantPointAttention_2430951489762
// MI455X (gfx1250) — matrix operand order repaired and checked by reading, not compiled, not run on hardware
//
#include <hip/hip_runtime.h>
#include <hip/hip_bf16.h>

// ---------------- problem constants ----------------
#define B_   2
#define N_   2048
#define CS   384
#define CC   256
#define CZ   128
#define H_   12
#define CH   16
#define PQK  4
#define PV   8
#define BQ   32
#define BK   128
#define NB   64
#define TOK  (B_*N_)          // 4096 tokens
#define NPROJ 1152            // 192 q | 192 k | 192 v | 144 qp | 432 kvp

typedef __attribute__((ext_vector_type(16))) _Float16 v16h;
typedef __attribute__((ext_vector_type(8)))  float    v8f;

// ---------------- WMMA helpers (gfx1250, wave32) ----------------
__device__ inline v8f wmma32(v16h a, v16h b, v8f c) {
  return __builtin_amdgcn_wmma_f32_16x16x32_f16(false, a, false, b, (short)0, c,
                                                false, false);
}

// A fragment: 16x32 f16, row-major source with leading dim ld.
// lane: M = lane%16 ; VGPR v holds K pair (2v + (v>=4?8:0)) + 8*(lane/16)
__device__ inline v16h load_a(const _Float16* src, int ld) {
  const int lane = threadIdx.x & 31;
  const _Float16* p = src + (size_t)(lane & 15) * ld + (lane >> 4) * 8;
  v16h a;
#pragma unroll
  for (int v = 0; v < 8; ++v) {
    const int kb = 2 * v + (v >= 4 ? 8 : 0);
    a[2 * v]     = p[kb];
    a[2 * v + 1] = p[kb + 1];
  }
  return a;
}

// B fragment: 32x16 f16, row-major source [K][N]; lane holds column N=lane%16,
// element i holds K = (i + (i>=8?8:0)) + 8*(lane/16).
__device__ inline v16h load_b(const _Float16* src, int ld) {
  const int lane = threadIdx.x & 31;
  const _Float16* p = src + (lane & 15) + (size_t)((lane >> 4) * 8) * ld;
  v16h b;
#pragma unroll
  for (int i = 0; i < 16; ++i) b[i] = p[(size_t)(i + (i >= 8 ? 8 : 0)) * ld];
  return b;
}

// C/D: lane -> col = lane%16, VGPR r -> row = r + 8*(lane/16)
__device__ inline void store_d_f32(float* dst, int ld, v8f d) {
  const int lane = threadIdx.x & 31;
  const int col = lane & 15, half = lane >> 4;
#pragma unroll
  for (int r = 0; r < 8; ++r) dst[(size_t)(r + 8 * half) * ld + col] = d[r];
}
__device__ inline void store_d_f16(_Float16* dst, int ld, v8f d) {
  const int lane = threadIdx.x & 31;
  const int col = lane & 15, half = lane >> 4;
#pragma unroll
  for (int r = 0; r < 8; ++r)
    dst[(size_t)(r + 8 * half) * ld + col] = (_Float16)d[r];
}

__device__ inline float wave_sum(float v) {
#pragma unroll
  for (int o = 16; o > 0; o >>= 1) v += __shfl_xor(v, o, 32);
  return v;
}
__device__ inline float wave_max(float v) {
#pragma unroll
  for (int o = 16; o > 0; o >>= 1) v = fmaxf(v, __shfl_xor(v, o, 32));
  return v;
}

// ================== K0: weight conversion / packing to f16 ==================
__global__ __launch_bounds__(256) void k0_convert(
    const float* Wq, const float* Wk, const float* Wv, const float* Wqp,
    const float* Wkvp, const float* Wb, const float* Wdz, const float* Wout,
    const float* adw, _Float16* hadw, _Float16* wscat, _Float16* hwz,
    _Float16* hwout) {
  const int R1 = CC * 2 * CS;     // adaln_w      196608
  const int R2 = CS * NPROJ;      // packed proj  442368
  const int R3 = CZ * 64;         // [sqrt(1/3)Wb | pad | Wdz | pad]
  const int R4 = (H_ * 80) * CS;  // W_out        368640
  const int total = R1 + R2 + R3 + R4;
  for (int idx = blockIdx.x * blockDim.x + threadIdx.x; idx < total;
       idx += gridDim.x * blockDim.x) {
    if (idx < R1) { hadw[idx] = (_Float16)adw[idx]; continue; }
    int j = idx - R1;
    if (j < R2) {
      const int k = j / NPROJ, c = j % NPROJ;
      float v;
      if (c < 192)      v = Wq[k * 192 + c];
      else if (c < 384) v = Wk[k * 192 + c - 192];
      else if (c < 576) v = Wv[k * 192 + c - 384];
      else if (c < 720) v = Wqp[k * 144 + c - 576];
      else              v = Wkvp[k * 432 + c - 720];
      wscat[j] = (_Float16)v;
      continue;
    }
    j -= R2;
    if (j < R3) {
      const int k = j / 64, c = j % 64;
      float v = 0.f;
      if (c < 12)                 v = Wb[k * 12 + c] * 0.57735026919f;
      else if (c >= 16 && c < 48) v = Wdz[k * 32 + c - 16];
      hwz[j] = (_Float16)v;
      continue;
    }
    j -= R3;
    hwout[j] = (_Float16)Wout[j];
  }
}

// ================== K1: adaLN + all per-token projections (WMMA) ===========
__global__ __launch_bounds__(256) void k1_proj(const float* s, const float* cond,
                                               const float* adb,
                                               const _Float16* hadw,
                                               const _Float16* wscat,
                                               float* proj) {
  __shared__ float    snorm[16 * CS];   // 24 KB
  __shared__ _Float16 cond_h[16 * CC];  // 8 KB
  __shared__ _Float16 sA[16 * CS];      // 12 KB
  const int tid = threadIdx.x, lane = tid & 31, wave = tid >> 5;
  const int half = lane >> 4;
  const int tok0 = blockIdx.x * 16;

  // LayerNorm(s) (no affine), 2 rows per wave
  for (int r = wave * 2; r < wave * 2 + 2; ++r) {
    const float* sr = s + (size_t)(tok0 + r) * CS + lane * 12;
    float x[12], sm = 0.f, s2 = 0.f;
#pragma unroll
    for (int j = 0; j < 12; ++j) { x[j] = sr[j]; sm += x[j]; s2 += x[j] * x[j]; }
    sm = wave_sum(sm); s2 = wave_sum(s2);
    const float mean = sm * (1.f / CS);
    const float rinv = rsqrtf(s2 * (1.f / CS) - mean * mean + 1e-5f);
#pragma unroll
    for (int j = 0; j < 12; ++j)
      snorm[r * CS + lane * 12 + j] = (x[j] - mean) * rinv;
  }
  for (int e = tid; e < 16 * CC; e += 256) {
    const int r = e / CC, c = e % CC;
    cond_h[e] = (_Float16)cond[(size_t)(tok0 + r) * CC + c];
  }
  __syncthreads();

  // scale/shift = cond @ adaln_w + adaln_b ; sA = snorm*(1+scale)+shift
  for (int n = wave; n < 24; n += 8) {
    v8f accS = {}, accT = {};
    for (int kc = 0; kc < 8; ++kc) {
      v16h A  = load_a(cond_h + kc * 32, CC);
      v16h Bs = load_b(hadw + (size_t)(kc * 32) * 768 + n * 16, 768);
      v16h Bt = load_b(hadw + (size_t)(kc * 32) * 768 + 384 + n * 16, 768);
      accS = wmma32(A, Bs, accS);
      accT = wmma32(A, Bt, accT);
    }
    const int col = n * 16 + (lane & 15);
    const float bs = adb[col], bt = adb[384 + col];
#pragma unroll
    for (int r = 0; r < 8; ++r) {
      const int m = r + 8 * half;
      const float sc = accS[r] + bs, sh = accT[r] + bt;
      sA[m * CS + col] = (_Float16)(snorm[m * CS + col] * (1.f + sc) + sh);
    }
  }
  __syncthreads();

  // proj = sA @ [Wq|Wk|Wv|Wqp|Wkvp]   (16 x 1152, K=384)
  for (int n = wave; n < 72; n += 8) {
    v8f acc = {};
    for (int kc = 0; kc < 12; ++kc)
      acc = wmma32(load_a(sA + kc * 32, CS),
                   load_b(wscat + (size_t)(kc * 32) * NPROJ + n * 16, NPROJ),
                   acc);
    store_d_f32(proj + (size_t)tok0 * NPROJ + n * 16, NPROJ, acc);
  }
}

// ======== K1b: per-token frame transforms + packed f16 attention feats =====
__global__ __launch_bounds__(256) void k1b_points(
    const float* proj, const float* rots, const float* trans, const float* hwts,
    _Float16* qcat, _Float16* kcat, _Float16* vh, _Float16* vpt, float* sqq,
    float* sqk) {
  __shared__ float Rl[9], trl[3], hwl[12], sqql[12], sqkl[12];
  const int tid = threadIdx.x;
  const size_t tok = blockIdx.x;
  const float* pr = proj + tok * NPROJ;
  if (tid < 9) Rl[tid] = rots[tok * 9 + tid];
  if (tid < 3) trl[tid] = trans[tok * 3 + tid];
  if (tid < 12) {
    // softplus(head_weight) * sqrt(1/(3*PQK*9/2)) = softplus * sqrt(1/54)
    hwl[tid]  = log1pf(__expf(hwts[tid])) * 0.13608276348795434f;
    sqql[tid] = 0.f;
    sqkl[tid] = 0.f;
  }
  __syncthreads();
  const float S1 = 0.14433756729740643f;  // sqrt(1/(3*CH))
  if (tid < 192) {
    const int h = tid >> 4, c = tid & 15;
    qcat[tok * 384 + h * 32 + c] = (_Float16)(pr[tid] * S1);
    kcat[tok * 384 + h * 32 + c] = (_Float16)pr[192 + tid];
    vh[tok * 192 + tid]          = (_Float16)pr[384 + tid];
  } else if (tid >= 208) {
    const int e = tid - 208;  // 48 pad slots: cols 28..31 of each head block
    const int h = e >> 2, sl = 28 + (e & 3);
    qcat[tok * 384 + h * 32 + sl] = (_Float16)0.f;
    kcat[tok * 384 + h * 32 + sl] = (_Float16)0.f;
  }
  if (tid < 48) {  // q_pts: R@p + t, scale by hw[h]
    const int h = tid >> 2, pp = tid & 3;
    const float* v = pr + 576 + tid * 3;
    const float r0 = Rl[0] * v[0] + Rl[1] * v[1] + Rl[2] * v[2] + trl[0];
    const float r1 = Rl[3] * v[0] + Rl[4] * v[1] + Rl[5] * v[2] + trl[1];
    const float r2 = Rl[6] * v[0] + Rl[7] * v[1] + Rl[8] * v[2] + trl[2];
    const float hw = hwl[h];
    qcat[tok * 384 + h * 32 + 16 + pp * 3 + 0] = (_Float16)(r0 * hw);
    qcat[tok * 384 + h * 32 + 16 + pp * 3 + 1] = (_Float16)(r1 * hw);
    qcat[tok * 384 + h * 32 + 16 + pp * 3 + 2] = (_Float16)(r2 * hw);
    atomicAdd(&sqql[h], 0.5f * hw * (r0 * r0 + r1 * r1 + r2 * r2));
  }
  if (tid >= 64 && tid < 208) {  // kv_pts (144 points)
    const int p = tid - 64;
    const int h = p / 12, pp = p % 12;
    const float* v = pr + 720 + p * 3;
    const float r0 = Rl[0] * v[0] + Rl[1] * v[1] + Rl[2] * v[2] + trl[0];
    const float r1 = Rl[3] * v[0] + Rl[4] * v[1] + Rl[5] * v[2] + trl[1];
    const float r2 = Rl[6] * v[0] + Rl[7] * v[1] + Rl[8] * v[2] + trl[2];
    if (pp < PQK) {
      kcat[tok * 384 + h * 32 + 16 + pp * 3 + 0] = (_Float16)r0;
      kcat[tok * 384 + h * 32 + 16 + pp * 3 + 1] = (_Float16)r1;
      kcat[tok * 384 + h * 32 + 16 + pp * 3 + 2] = (_Float16)r2;
      atomicAdd(&sqkl[h], 0.5f * hwl[h] * (r0 * r0 + r1 * r1 + r2 * r2));
    } else {
      vpt[tok * 288 + h * 24 + (pp - PQK) * 3 + 0] = (_Float16)r0;
      vpt[tok * 288 + h * 24 + (pp - PQK) * 3 + 1] = (_Float16)r1;
      vpt[tok * 288 + h * 24 + (pp - PQK) * 3 + 2] = (_Float16)r2;
    }
  }
  __syncthreads();
  if (tid < 12) {
    sqq[tok * 12 + tid] = sqql[tid];
    sqk[tok * 12 + tid] = sqkl[tid];
  }
}

// ======== K3: fused IPA per query — z is read from HBM exactly once ========
__global__ __launch_bounds__(256) void k3_attn(
    const float* z, const float* lnzg, const float* lnzb, const _Float16* hwz,
    const _Float16* qcat, const _Float16* kcat, const _Float16* vh,
    const _Float16* vpt, const float* sqq, const float* sqk, const float* smask,
    const int* keyidx, const float* rots, const float* trans, _Float16* ocat) {
  __shared__ _Float16 zn[BK * CZ];     // 32 KB  ln(z) f16
  __shared__ float    zbuf[BK * 16];   // 8 KB   sqrt(1/3)*zn@W_b
  __shared__ float    Lb[H_ * BK];     // 6 KB   logits
  __shared__ _Float16 ah[16 * BK];     // 4 KB   softmax(a), rows 12..15 = 0
  __shared__ _Float16 th[16 * BK];     // 4 KB   t = a@zn
  __shared__ _Float16 qc[H_ * 32];
  __shared__ float    gz[CZ], bz[CZ];
  __shared__ float    sqql[H_];
  __shared__ float    ol[H_ * CH];
  __shared__ float    optg[H_ * PV * 3];
  __shared__ float    opr[16 * 32];
  __shared__ float    Rq[9], trq[3], mqs[1];
  __shared__ int      kst[1];

  const int tid = threadIdx.x, lane = tid & 31, wave = tid >> 5;
  const int g  = blockIdx.x;            // flat (b, nb, q)
  const int b  = g / (NB * BQ);
  const int nb = (g / BQ) % NB;
  const int q  = g % BQ;
  const int bN = b * N_;
  const int tokq = bN + nb * BQ + q;

  if (tid == 0) { kst[0] = keyidx[nb * BK]; mqs[0] = smask[tokq]; }
  if (tid < 9) Rq[tid] = rots[(size_t)tokq * 9 + tid];
  if (tid < 3) trq[tid] = trans[(size_t)tokq * 3 + tid];
  if (tid < CZ) { gz[tid] = lnzg[tid]; bz[tid] = lnzb[tid]; }
  if (tid < H_) sqql[tid] = sqq[(size_t)tokq * 12 + tid];
  for (int e = tid; e < H_ * 32; e += 256) qc[e] = qcat[(size_t)tokq * 384 + e];
  for (int e = tid; e < 4 * BK; e += 256)
    ah[(12 + e / BK) * BK + (e % BK)] = (_Float16)0.f;  // zero pad rows
  __syncthreads();

  // 1) stream z (contiguous 128x128 f32), LN each key row -> f16 LDS
  const size_t zbase = (size_t)g * (BK * CZ);
  for (int r = 0; r < 16; ++r) {
    const int k = wave * 16 + r;
    const float4 x =
        *(const float4*)(z + zbase + (size_t)k * CZ + lane * 4);
    float sm = x.x + x.y + x.z + x.w;
    float s2 = x.x * x.x + x.y * x.y + x.z * x.z + x.w * x.w;
    sm = wave_sum(sm); s2 = wave_sum(s2);
    const float mean = sm * (1.f / CZ);
    const float rinv = rsqrtf(s2 * (1.f / CZ) - mean * mean + 1e-5f);
    const int c = lane * 4;
    zn[k * CZ + c + 0] = (_Float16)((x.x - mean) * rinv * gz[c + 0] + bz[c + 0]);
    zn[k * CZ + c + 1] = (_Float16)((x.y - mean) * rinv * gz[c + 1] + bz[c + 1]);
    zn[k * CZ + c + 2] = (_Float16)((x.z - mean) * rinv * gz[c + 2] + bz[c + 2]);
    zn[k * CZ + c + 3] = (_Float16)((x.w - mean) * rinv * gz[c + 3] + bz[c + 3]);
  }
  __syncthreads();

  // 2) zb = zn @ [sqrt(1/3)*W_b | zero-pad]  (M=128 keys, N=16, K=128)
  {
    const int k0 = wave * 16;
    v8f acc = {};
    for (int kc = 0; kc < 4; ++kc)
      acc = wmma32(load_a(zn + k0 * CZ + kc * 32, CZ),
                   load_b(hwz + (size_t)(kc * 32) * 64, 64), acc);
    store_d_f32(zbuf + k0 * 16, 16, acc);
  }
  __syncthreads();

  // 3) logits: qk*sqrt(1/48) + hw*qpt.kpt (packed K=32 dot) + zb - sq - mask
  const int kstart = kst[0];
  const float mq = mqs[0];
  for (int e = tid; e < H_ * BK; e += 256) {
    const int h = e / BK, k = e % BK;
    const size_t tk = (size_t)(bN + kstart + k);
    const _Float16* kp = kcat + tk * 384 + h * 32;
    float acc = 0.f;
#pragma unroll
    for (int j = 0; j < 32; ++j) acc += (float)qc[h * 32 + j] * (float)kp[j];
    const float mk = smask[tk];
    Lb[e] = acc + zbuf[k * 16 + h] - sqql[h] - sqk[tk * 12 + h] +
            (mq * mk - 1.f) * 100000.f;
  }
  __syncthreads();

  // 4) softmax over k per head (wave32 reductions)
  for (int h = wave; h < H_; h += 8) {
    float v0 = Lb[h * BK + lane], v1 = Lb[h * BK + lane + 32];
    float v2 = Lb[h * BK + lane + 64], v3 = Lb[h * BK + lane + 96];
    float m = wave_max(fmaxf(fmaxf(v0, v1), fmaxf(v2, v3)));
    v0 = __expf(v0 - m); v1 = __expf(v1 - m);
    v2 = __expf(v2 - m); v3 = __expf(v3 - m);
    const float rs = 1.f / wave_sum(v0 + v1 + v2 + v3);
    ah[h * BK + lane]      = (_Float16)(v0 * rs);
    ah[h * BK + lane + 32] = (_Float16)(v1 * rs);
    ah[h * BK + lane + 64] = (_Float16)(v2 * rs);
    ah[h * BK + lane + 96] = (_Float16)(v3 * rs);
  }
  __syncthreads();

  // 5) t = a @ zn  (16 x 128, K=128) — WMMA, one N-tile per wave
  {
    const int n0 = wave * 16;
    v8f acc = {};
    for (int kc = 0; kc < 4; ++kc)
      acc = wmma32(load_a(ah + kc * 32, BK),
                   load_b(zn + (size_t)(kc * 32) * CZ + n0, CZ), acc);
    store_d_f16(th + n0, BK, acc);
  }
  // 6) o = a@v, o_pt = a@v_pts  (small per-query reductions; operands in L2)
  for (int e = tid; e < 480; e += 256) {
    if (e < 192) {
      const int h = e >> 4, c = e & 15;
      float acc = 0.f;
      for (int k = 0; k < BK; ++k)
        acc += (float)ah[h * BK + k] *
               (float)vh[(size_t)(bN + kstart + k) * 192 + h * 16 + c];
      ol[e] = acc;
    } else {
      const int e2 = e - 192, h = e2 / 24, j = e2 % 24;
      float acc = 0.f;
      for (int k = 0; k < BK; ++k)
        acc += (float)ah[h * BK + k] *
               (float)vpt[(size_t)(bN + kstart + k) * 288 + h * 24 + j];
      optg[e2] = acc;
    }
  }
  __syncthreads();

  // 7) o_pair = t @ W_down_z  (16 x 32, K=128)
  if (wave < 2) {
    v8f acc = {};
    for (int kc = 0; kc < 4; ++kc)
      acc = wmma32(load_a(th + kc * 32, BK),
                   load_b(hwz + (size_t)(kc * 32) * 64 + 16 + wave * 16, 64),
                   acc);
    store_d_f32(opr + wave * 16, 32, acc);
  }
  __syncthreads();

  // 8) assemble ocat row: [o(192) | o_pt local(288) | norms(96) | o_pair(384)]
  _Float16* oc = ocat + (size_t)tokq * 960;
  if (tid < 192) oc[tid] = (_Float16)ol[tid];
  if (tid < 96) {
    const int h = tid >> 3, p = tid & 7;
    const float d0 = optg[h * 24 + p * 3 + 0] - trq[0];
    const float d1 = optg[h * 24 + p * 3 + 1] - trq[1];
    const float d2 = optg[h * 24 + p * 3 + 2] - trq[2];
    const float l0 = Rq[0] * d0 + Rq[3] * d1 + Rq[6] * d2;  // R^T
    const float l1 = Rq[1] * d0 + Rq[4] * d1 + Rq[7] * d2;
    const float l2 = Rq[2] * d0 + Rq[5] * d1 + Rq[8] * d2;
    oc[192 + h * 24 + p * 3 + 0] = (_Float16)l0;
    oc[192 + h * 24 + p * 3 + 1] = (_Float16)l1;
    oc[192 + h * 24 + p * 3 + 2] = (_Float16)l2;
    oc[480 + h * 8 + p] = (_Float16)sqrtf(l0 * l0 + l1 * l1 + l2 * l2 + 1e-8f);
  }
  for (int e = tid; e < 384; e += 256) oc[576 + e] = (_Float16)opr[e];
}

// ================== K4: out = ocat @ W_out (WMMA) ==========================
__global__ __launch_bounds__(256) void k4_out(const _Float16* ocat,
                                              const _Float16* hwout,
                                              float* outp) {
  const int wave = threadIdx.x >> 5;
  const int tok0 = blockIdx.x * 16;
  for (int n = wave; n < 24; n += 8) {
    v8f acc = {};
    for (int kc = 0; kc < 30; ++kc)
      acc = wmma32(load_a(ocat + (size_t)tok0 * 960 + kc * 32, 960),
                   load_b(hwout + (size_t)(kc * 32) * CS + n * 16, CS), acc);
    store_d_f32(outp + (size_t)tok0 * CS + n * 16, CS, acc);
  }
}

// ================== host launcher ==========================================
extern "C" void kernel_launch(void* const* d_in, const int* in_sizes, int n_in,
                              void* d_out, int out_size, void* d_ws,
                              size_t ws_size, hipStream_t stream) {
  (void)in_sizes; (void)n_in; (void)out_size; (void)ws_size;
  const float* s      = (const float*)d_in[0];
  const float* cond   = (const float*)d_in[1];
  const float* z      = (const float*)d_in[2];
  const float* trans  = (const float*)d_in[3];
  const float* rots   = (const float*)d_in[4];
  const float* smask  = (const float*)d_in[5];
  const int*   keyidx = (const int*)d_in[6];
  const float* Wq     = (const float*)d_in[7];
  const float* Wk     = (const float*)d_in[8];
  const float* Wv     = (const float*)d_in[9];
  const float* Wqp    = (const float*)d_in[10];
  const float* Wkvp   = (const float*)d_in[11];
  const float* Wb     = (const float*)d_in[12];
  const float* Wdz    = (const float*)d_in[13];
  const float* hwts   = (const float*)d_in[14];
  const float* Wout   = (const float*)d_in[15];
  const float* adw    = (const float*)d_in[16];
  const float* adb    = (const float*)d_in[17];
  const float* lnzg   = (const float*)d_in[18];
  const float* lnzb   = (const float*)d_in[19];

  char* ws = (char*)d_ws;
  size_t off = 0;
  _Float16* hadw  = (_Float16*)(ws + off); off += (size_t)CC * 768 * 2;
  _Float16* wscat = (_Float16*)(ws + off); off += (size_t)CS * NPROJ * 2;
  _Float16* hwz   = (_Float16*)(ws + off); off += (size_t)CZ * 64 * 2;
  _Float16* hwout = (_Float16*)(ws + off); off += (size_t)960 * CS * 2;
  float*    proj  = (float*)(ws + off);    off += (size_t)TOK * NPROJ * 4;
  _Float16* qcat  = (_Float16*)(ws + off); off += (size_t)TOK * 384 * 2;
  _Float16* kcat  = (_Float16*)(ws + off); off += (size_t)TOK * 384 * 2;
  _Float16* vhp   = (_Float16*)(ws + off); off += (size_t)TOK * 192 * 2;
  _Float16* vptp  = (_Float16*)(ws + off); off += (size_t)TOK * 288 * 2;
  float*    sqq   = (float*)(ws + off);    off += (size_t)TOK * 12 * 4;
  float*    sqk   = (float*)(ws + off);    off += (size_t)TOK * 12 * 4;
  _Float16* ocat  = (_Float16*)(ws + off); off += (size_t)TOK * 960 * 2;
  // total ws use ~= 39.4 MB

  k0_convert<<<512, 256, 0, stream>>>(Wq, Wk, Wv, Wqp, Wkvp, Wb, Wdz, Wout,
                                      adw, hadw, wscat, hwz, hwout);
  k1_proj<<<TOK / 16, 256, 0, stream>>>(s, cond, adb, hadw, wscat, proj);
  k1b_points<<<TOK, 256, 0, stream>>>(proj, rots, trans, hwts, qcat, kcat, vhp,
                                      vptp, sqq, sqk);
  k3_attn<<<TOK, 256, 0, stream>>>(z, lnzg, lnzb, hwz, qcat, kcat, vhp, vptp,
                                   sqq, sqk, smask, keyidx, rots, trans, ocat);
  k4_out<<<TOK / 16, 256, 0, stream>>>(ocat, hwout, (float*)d_out);
}
